// EvolveGCN_9019431321777
// MI455X (gfx1250) — matrix operand order repaired and checked by reading, not compiled, not run on hardware
//
#include <hip/hip_runtime.h>

#define N_NODES 100000
#define N_EDGES 3200000
#define DIM     64
#define HID     4096   // DIM*DIM

typedef __attribute__((ext_vector_type(16))) __bf16 v16bf;
typedef __attribute__((ext_vector_type(8)))  float  v8f;

// ---------------------------------------------------------------------------
// Degree accumulation: one thread per edge, f32 atomics (L2-resident targets).
// ---------------------------------------------------------------------------
__global__ void degree_kernel(const int* __restrict__ src, const int* __restrict__ dst,
                              float* __restrict__ outd, float* __restrict__ ind) {
    int e = blockIdx.x * blockDim.x + threadIdx.x;
    if (e < N_EDGES) {
        atomicAdd(&outd[src[e]], 1.0f);
        atomicAdd(&ind[dst[e]], 1.0f);
    }
}

__global__ void norm_kernel(float* __restrict__ a, float* __restrict__ b) {
    int i = blockIdx.x * blockDim.x + threadIdx.x;
    if (i < N_NODES) {
        a[i] = rsqrtf(fmaxf(a[i], 1.0f));
        b[i] = rsqrtf(fmaxf(b[i], 1.0f));
    }
}

// ---------------------------------------------------------------------------
// GRU weight evolution. HBM-bound: 402 MB of W_ih/W_hh read exactly once.
// One wave32 per output element j; each wave computes the 6 needed weight rows
// dotted against BOTH GRU instances' (x, h) vectors -> 12 accumulators.
// ---------------------------------------------------------------------------
__device__ __forceinline__ float dot4(float4 a, float4 b) {
    return fmaf(a.x, b.x, fmaf(a.y, b.y, fmaf(a.z, b.z, a.w * b.w)));
}
__device__ __forceinline__ float sigmoidf(float x) { return 1.0f / (1.0f + expf(-x)); }

__global__ void gru_kernel(const float* __restrict__ Wih, const float* __restrict__ Whh,
                           const float* __restrict__ bih, const float* __restrict__ bhh,
                           const float* __restrict__ x1, const float* __restrict__ h1,
                           const float* __restrict__ x2, const float* __restrict__ h2,
                           float* __restrict__ w1, float* __restrict__ w2) {
    const int lane = threadIdx.x & 31;
    const int j    = (blockIdx.x * blockDim.x + threadIdx.x) >> 5;
    if (j >= HID) return;

    const float* ri0 = Wih + (size_t)j * HID;
    const float* ri1 = Wih + (size_t)(j +     HID) * HID;
    const float* ri2 = Wih + (size_t)(j + 2 * HID) * HID;
    const float* rh0 = Whh + (size_t)j * HID;
    const float* rh1 = Whh + (size_t)(j +     HID) * HID;
    const float* rh2 = Whh + (size_t)(j + 2 * HID) * HID;

    float s[12];
#pragma unroll
    for (int i = 0; i < 12; ++i) s[i] = 0.0f;

    for (int k = lane * 4; k < HID; k += 32 * 4) {
        float4 a0 = *(const float4*)(ri0 + k);
        float4 a1 = *(const float4*)(ri1 + k);
        float4 a2 = *(const float4*)(ri2 + k);
        float4 b0 = *(const float4*)(rh0 + k);
        float4 b1 = *(const float4*)(rh1 + k);
        float4 b2 = *(const float4*)(rh2 + k);
        float4 vx1 = *(const float4*)(x1 + k);
        float4 vx2 = *(const float4*)(x2 + k);
        float4 vh1 = *(const float4*)(h1 + k);
        float4 vh2 = *(const float4*)(h2 + k);
        s[0]  += dot4(a0, vx1); s[1]  += dot4(a1, vx1); s[2]  += dot4(a2, vx1);
        s[3]  += dot4(a0, vx2); s[4]  += dot4(a1, vx2); s[5]  += dot4(a2, vx2);
        s[6]  += dot4(b0, vh1); s[7]  += dot4(b1, vh1); s[8]  += dot4(b2, vh1);
        s[9]  += dot4(b0, vh2); s[10] += dot4(b1, vh2); s[11] += dot4(b2, vh2);
    }
    // wave32 xor reduction
#pragma unroll
    for (int i = 0; i < 12; ++i) {
#pragma unroll
        for (int m = 16; m > 0; m >>= 1) s[i] += __shfl_xor(s[i], m, 32);
    }

    if (lane == 0) {
        float bir = bih[j], biz = bih[j + HID], bin = bih[j + 2 * HID];
        float bhr = bhh[j], bhz = bhh[j + HID], bhn = bhh[j + 2 * HID];
        {   // instance 1: x = prev_gc1, h = gc1_weight(flat)
            float r = sigmoidf((s[0] + bir) + (s[6] + bhr));
            float z = sigmoidf((s[1] + biz) + (s[7] + bhz));
            float n = tanhf((s[2] + bin) + r * (s[8] + bhn));
            w1[j] = (1.0f - z) * n + z * h1[j];
        }
        {   // instance 2
            float r = sigmoidf((s[3] + bir) + (s[9]  + bhr));
            float z = sigmoidf((s[4] + biz) + (s[10] + bhz));
            float n = tanhf((s[5] + bin) + r * (s[11] + bhn));
            w2[j] = (1.0f - z) * n + z * h2[j];
        }
    }
}

// ---------------------------------------------------------------------------
// y[node,:] = x[node,:] * nrm[node]   (works in-place, float4 vectorized)
// ---------------------------------------------------------------------------
__global__ void scale_kernel(const float* __restrict__ x, const float* __restrict__ nrm,
                             float* __restrict__ y) {
    int t = blockIdx.x * blockDim.x + threadIdx.x;
    if (t < N_NODES * (DIM / 4)) {
        float s = nrm[t / (DIM / 4)];
        float4 v = ((const float4*)x)[t];
        v.x *= s; v.y *= s; v.z *= s; v.w *= s;
        ((float4*)y)[t] = v;
    }
}

// ---------------------------------------------------------------------------
// Edge scatter: one wave32 per edge, float2 per lane (64 floats/edge).
// Gather + atomic scatter both hit the 25.6 MB feature tables (L2-resident).
// ---------------------------------------------------------------------------
__global__ void scatter_kernel(const int* __restrict__ src, const int* __restrict__ dst,
                               const float* __restrict__ xs, float* __restrict__ agg) {
    const int wid  = (blockIdx.x * blockDim.x + threadIdx.x) >> 5;
    const int lane = threadIdx.x & 31;
    if (wid < N_EDGES) {
        int s = __builtin_amdgcn_readfirstlane(src[wid]);   // wave-uniform -> SGPR base
        int d = __builtin_amdgcn_readfirstlane(dst[wid]);
        float2 v = ((const float2*)(xs + (size_t)s * DIM))[lane];
        float* p = agg + (size_t)d * DIM + lane * 2;
        atomicAdd(p,     v.x);
        atomicAdd(p + 1, v.y);
    }
}

// ---------------------------------------------------------------------------
// GCN GEMM: out = act( (agg * in_norm) @ W + b ) via v_wmma_f32_16x16x32_bf16.
// Wave owns a 16(M)x64(N) stripe: 2 K-steps x 4 N-tiles = 8 WMMAs.
//
// W (64x64) is pre-swizzled into B-fragment order in LDS:
//   Wl[kk][nt][hi][r][i]  (i fastest; 16 bf16 = 32 contiguous bytes)
//   where fragment element i of lane L (hi=L>>4, r=L&15) for tile (kk,nt)
//   corresponds to W[k][n] with k = kk*32 + 8*hi + (i&7) + 16*(i>>3), n = nt*16 + r.
// Each lane then fetches a whole B fragment with one aligned 32-byte DS read
// (2x ds_load_b128) instead of 16x ds_load_u16.
//
// A fragment follows ISA 16-bit A layout: element i ->
//   K = kk*32 + 8*hi + (i&7) + (i>=8 ? 16 : 0).
// C/D layout: VGPR e -> row e + 8*hi, column = 16*ntile + (lane&15).
// 100000 % 16 == 0, so all active tiles are full and EXEC is all-ones at the
// WMMA (ISA requirement); inactive waves exit via a wave-uniform branch.
// ---------------------------------------------------------------------------
__global__ void gcn_gemm_kernel(const float* __restrict__ agg, const float* __restrict__ innorm,
                                const float* __restrict__ W, const float* __restrict__ bias,
                                float* __restrict__ out, int relu) {
    __shared__ __align__(32) __bf16 Wl[DIM * DIM];   // 8 KB, fragment-ordered
    for (int idx = threadIdx.x; idx < DIM * DIM; idx += blockDim.x) {
        const int i  =  idx        & 15;
        const int r  = (idx >> 4)  & 15;
        const int hi = (idx >> 8)  & 1;
        const int nt = (idx >> 9)  & 3;
        const int kk = (idx >> 11) & 1;
        const int k  = kk * 32 + 8 * hi + (i & 7) + ((i >> 3) << 4);
        const int n  = nt * 16 + r;
        Wl[idx] = (__bf16)W[k * DIM + n];
    }
    __syncthreads();

    const int lane = threadIdx.x & 31;
    const int wave = threadIdx.x >> 5;
    const int tile = blockIdx.x * 8 + wave;          // 16-row M tile
    if (tile * 16 >= N_NODES) return;

    const int hi = lane >> 4;
    const int r  = lane & 15;
    const int node  = tile * 16 + r;
    const float scl = innorm[node];
    const float* arow = agg + (size_t)node * DIM;

    v8f acc[4] = {};
#pragma unroll
    for (int kk = 0; kk < 2; ++kk) {
        const int kbase = kk * 32 + 8 * hi;
        float4 f0 = *(const float4*)(arow + kbase);
        float4 f1 = *(const float4*)(arow + kbase + 4);
        float4 f2 = *(const float4*)(arow + kbase + 16);
        float4 f3 = *(const float4*)(arow + kbase + 20);
        float fv[16] = { f0.x, f0.y, f0.z, f0.w, f1.x, f1.y, f1.z, f1.w,
                         f2.x, f2.y, f2.z, f2.w, f3.x, f3.y, f3.z, f3.w };
        v16bf a;
#pragma unroll
        for (int i = 0; i < 16; ++i) a[i] = (__bf16)(fv[i] * scl);

#pragma unroll
        for (int nt = 0; nt < 4; ++nt) {
            // whole fragment in one aligned 32-byte LDS read
            const v16bf b = *(const v16bf*)(Wl + ((((kk * 4 + nt) * 2 + hi) * 16 + r) * 16));
            acc[nt] = __builtin_amdgcn_wmma_f32_16x16x32_bf16(
                false, a, false, b, (short)0, acc[nt], false, false);
        }
    }

#pragma unroll
    for (int nt = 0; nt < 4; ++nt) {
        const float bv = bias[nt * 16 + r];
#pragma unroll
        for (int e = 0; e < 8; ++e) {
            const int M = e + 8 * hi;
            float v = acc[nt][e] + bv;
            if (relu) v = fmaxf(v, 0.0f);
            out[(size_t)(tile * 16 + M) * DIM + nt * 16 + r] = v;
        }
    }
}

// ---------------------------------------------------------------------------
extern "C" void kernel_launch(void* const* d_in, const int* in_sizes, int n_in,
                              void* d_out, int out_size, void* d_ws, size_t ws_size,
                              hipStream_t stream) {
    const float* node_emb = (const float*)d_in[0];
    const float* gc1_w    = (const float*)d_in[1];
    const float* gc2_w    = (const float*)d_in[2];
    const float* gc1_b    = (const float*)d_in[3];
    const float* gc2_b    = (const float*)d_in[4];
    const float* prev1    = (const float*)d_in[5];
    const float* prev2    = (const float*)d_in[6];
    const float* Wih      = (const float*)d_in[7];
    const float* Whh      = (const float*)d_in[8];
    const float* bih      = (const float*)d_in[9];
    const float* bhh      = (const float*)d_in[10];
    const int*   src      = (const int*)d_in[11];
    const int*   dst      = (const int*)d_in[12];
    float* out = (float*)d_out;

    // workspace layout
    float* out_norm = (float*)d_ws;                       // N_NODES
    float* in_norm  = out_norm + N_NODES;                 // N_NODES
    float* w1       = in_norm + N_NODES;                  // HID
    float* w2       = w1 + HID;                           // HID
    float* bufA     = w2 + HID;                           // N_NODES*DIM
    float* bufB     = bufA + (size_t)N_NODES * DIM;       // N_NODES*DIM

    const int thr = 256;

    // degrees + norms
    hipMemsetAsync(out_norm, 0, 2 * N_NODES * sizeof(float), stream);
    degree_kernel<<<(N_EDGES + thr - 1) / thr, thr, 0, stream>>>(src, dst, out_norm, in_norm);
    norm_kernel<<<(N_NODES + thr - 1) / thr, thr, 0, stream>>>(out_norm, in_norm);

    // GRU weight evolution (one wave per output element, 4096 waves)
    gru_kernel<<<(HID * 32) / thr, thr, 0, stream>>>(Wih, Whh, bih, bhh,
                                                     prev1, gc1_w, prev2, gc2_w, w1, w2);

    const int scale_blocks   = (N_NODES * (DIM / 4) + thr - 1) / thr;
    const int scatter_blocks = (N_EDGES * 32) / thr;         // one wave per edge
    const int gemm_blocks    = ((N_NODES + 15) / 16 + 7) / 8;

    // layer 1
    scale_kernel<<<scale_blocks, thr, 0, stream>>>(node_emb, out_norm, bufA);
    hipMemsetAsync(bufB, 0, (size_t)N_NODES * DIM * sizeof(float), stream);
    scatter_kernel<<<scatter_blocks, thr, 0, stream>>>(src, dst, bufA, bufB);
    gcn_gemm_kernel<<<gemm_blocks, thr, 0, stream>>>(bufB, in_norm, w1, gc1_b, bufA, 1);

    // layer 2
    scale_kernel<<<scale_blocks, thr, 0, stream>>>(bufA, out_norm, bufA);
    hipMemsetAsync(bufB, 0, (size_t)N_NODES * DIM * sizeof(float), stream);
    scatter_kernel<<<scatter_blocks, thr, 0, stream>>>(src, dst, bufA, bufB);
    gcn_gemm_kernel<<<gemm_blocks, thr, 0, stream>>>(bufB, in_norm, w2, gc2_b, out, 0);
}
